// FlowMatchingModule_53601191854201
// MI455X (gfx1250) — matrix operand order repaired and checked by reading, not compiled, not run on hardware
//
#include <hip/hip_runtime.h>
#include <hip/hip_bf16.h>

typedef __attribute__((ext_vector_type(16))) __bf16 v16bf;
typedef __attribute__((ext_vector_type(8)))  __bf16 v8bf;
typedef __attribute__((ext_vector_type(8)))  float  v8f;

union AF { v16bf v; v8bf h[2]; };

#define WMMA_BF16(a, b, c) \
  __builtin_amdgcn_wmma_f32_16x16x32_bf16(false, (a), false, (b), (short)0, (c), false, false)

constexpr int Bn = 32, Tn = 2048, Dn = 512, Hn = 1024, STEPSn = 8;
constexpr int MB  = 32;             // rows per workgroup tile
constexpr int SX  = 516;            // f32 LDS row stride (516 dwords % 64 == 4 -> conflict-free)
constexpr int SXB = 520;            // bf16 LDS row stride for x (1040 B = 260 dw % 64 == 4)
constexpr int SH  = 1032;           // bf16 LDS row stride for h (2064 B = 516 dw % 64 == 4)
constexpr int KT1 = Dn / 32;        // 16 k-tiles for GEMM1
constexpr int KT2 = Hn / 32;        // 32 k-tiles for GEMM2

// ---------------------------------------------------------------------------
// Pack a (K x N) f32 row-major weight matrix into bf16 B-fragment layout:
// out[((nt*KT + kt)*32 + lane)*16 + j] = W[kt*32 + (j>=8 ? 16 : 0) + (lane>=16 ? 8 : 0) + (j&7)][nt*16 + (lane&15)]
// so each lane's 16 bf16 values for one 32x16 K-tile are 32 contiguous bytes.
// ---------------------------------------------------------------------------
__global__ void pack_w_kernel(const float* __restrict__ W, __bf16* __restrict__ out,
                              int K, int N) {
  int gid = blockIdx.x * 256 + threadIdx.x;
  if (gid >= K * N) return;
  int KT   = K >> 5;
  int j    = gid & 15;
  int lane = (gid >> 4) & 31;
  int tile = gid >> 9;
  int nt   = tile / KT;
  int kt   = tile - nt * KT;
  int k = (kt << 5) + ((j >> 3) << 4) + ((lane >> 4) << 3) + (j & 7);
  int n = (nt << 4) + (lane & 15);
  out[gid] = (__bf16)W[(size_t)k * N + n];
}

// bias1[s][h] = b1[h] + sum_e (t_s*wt[e] + bt[e]) * W1[512+e][h],  t_s = (8-s)/8
__global__ void bias_kernel(const float* __restrict__ W1, const float* __restrict__ b1,
                            const float* __restrict__ wt, const float* __restrict__ bt,
                            float* __restrict__ bias1) {
  int gid = blockIdx.x * 256 + threadIdx.x;
  if (gid >= STEPSn * Hn) return;
  int s = gid >> 10;
  int h = gid & (Hn - 1);
  float t = (float)(STEPSn - s) / (float)STEPSn;
  float acc = b1[h];
  for (int e = 0; e < 32; ++e)
    acc += (t * wt[e] + bt[e]) * W1[(size_t)(Dn + e) * Hn + h];
  bias1[gid] = acc;
}

// accum[0..7] = 0 (per-step loss sums), accum[8] = sum(lengths)*D (denominator)
__global__ void init_kernel(const int* __restrict__ lengths, float* __restrict__ accum) {
  int tid = threadIdx.x;
  if (tid < STEPSn) accum[tid] = 0.0f;
  if (tid == STEPSn) {
    long s = 0;
    for (int i = 0; i < Bn; ++i) s += lengths[i];
    accum[STEPSn] = (float)s * (float)Dn;
  }
}

__global__ void final_kernel(const float* __restrict__ accum, float* __restrict__ out) {
  float denom = fmaxf(accum[STEPSn], 1.0f);
  float s = 0.0f;
  for (int i = 0; i < STEPSn; ++i) s += accum[i];
  out[0] = s / denom * (1.0f / (float)STEPSn);
}

// ---------------------------------------------------------------------------
// Fused 8-step flow: x tile resident in LDS across all steps.
// Per wave GEMM tiling: 2 M-tiles x 4 N-tiles = 8 independent accumulators,
// so dependent WMMA pairs are 8 issues apart (hazard slots filled with real work)
// and each A-fragment ds_load feeds 4 WMMAs.
// ---------------------------------------------------------------------------
__global__ __launch_bounds__(256, 1)
void flow_main(const float* __restrict__ x0, const float* __restrict__ teacher,
               const int* __restrict__ lengths,
               const __bf16* __restrict__ pW1, const __bf16* __restrict__ pW2,
               const float* __restrict__ bias1, const float* __restrict__ b2g,
               float* __restrict__ accum) {
  __shared__ float  xf[MB * SX];     // x tile, f32 master copy
  __shared__ __bf16 xb[MB * SXB];    // x tile, bf16 for WMMA A-frags
  __shared__ __bf16 hb[MB * SH];     // hidden tile, bf16
  __shared__ float  lsum[STEPSn];

  const int tid  = threadIdx.x;
  const int lane = tid & 31;
  const int wv   = tid >> 5;
  const int half = lane >> 4;
  const int l15  = lane & 15;

  const int rowBase = blockIdx.x * MB;
  const int b  = rowBase >> 11;       // T = 2048
  const int t0 = rowBase & (Tn - 1);
  const int len = lengths[b];
  if (t0 >= len) return;              // uniform: tile fully masked, x not an output

  if (tid < STEPSn) lsum[tid] = 0.0f;

  const float* xg = x0      + (size_t)rowBase * Dn;
  const float* tg = teacher + (size_t)rowBase * Dn;

  for (int i = tid; i < MB * Dn; i += 256) {
    int m = i >> 9, c = i & (Dn - 1);
    xf[m * SX + c] = xg[i];
  }
  __syncthreads();

#pragma unroll 1
  for (int s = 0; s < STEPSn; ++s) {
    // --- cast x -> bf16 staging (reused by all 64 N-tiles of GEMM1) ---
    for (int i = tid; i < MB * Dn; i += 256) {
      int m = i >> 9, c = i & (Dn - 1);
      xb[m * SXB + c] = (__bf16)xf[m * SX + c];
    }
    __syncthreads();

    // --- GEMM1: h = relu(x @ W1x + bias_s) ; per wave: 2 passes of 2M x 4N ---
    const float* bs = bias1 + s * Hn;
#pragma unroll 1
    for (int p = 0; p < 2; ++p) {
      const int nt0 = wv * 8 + p * 4;
      v8f c0[4], c1[4];
      const __bf16* pB[4];
#pragma unroll
      for (int q = 0; q < 4; ++q) {
        const float bv = bs[(nt0 + q) * 16 + l15];
        c0[q] = (v8f){bv, bv, bv, bv, bv, bv, bv, bv};
        c1[q] = c0[q];
        pB[q] = pW1 + (size_t)(nt0 + q) * KT1 * 512 + lane * 16;
      }
#pragma unroll 4
      for (int kt = 0; kt < KT1; ++kt) {
        const int k0 = kt * 32;
        AF a0, a1;
        a0.h[0] = *(const v8bf*)&xb[l15 * SXB + k0 + half * 8];
        a0.h[1] = *(const v8bf*)&xb[l15 * SXB + k0 + 16 + half * 8];
        a1.h[0] = *(const v8bf*)&xb[(16 + l15) * SXB + k0 + half * 8];
        a1.h[1] = *(const v8bf*)&xb[(16 + l15) * SXB + k0 + 16 + half * 8];
        AF f[4];
#pragma unroll
        for (int q = 0; q < 4; ++q)
          f[q].v = *(const v16bf*)(pB[q] + (size_t)kt * 512);
#pragma unroll
        for (int q = 0; q < 4; ++q) {
          c0[q] = WMMA_BF16(a0.v, f[q].v, c0[q]);
          c1[q] = WMMA_BF16(a1.v, f[q].v, c1[q]);
        }
      }
#pragma unroll
      for (int q = 0; q < 4; ++q) {
        const int nc = (nt0 + q) * 16 + l15;
#pragma unroll
        for (int r = 0; r < 8; ++r) {
          const int m0 = 8 * half + r;
          hb[m0 * SH + nc]        = (__bf16)fmaxf(c0[q][r], 0.0f);
          hb[(16 + m0) * SH + nc] = (__bf16)fmaxf(c1[q][r], 0.0f);
        }
      }
    }
    __syncthreads();

    // --- GEMM2: v = h @ W2 ; x -= (v + b2)/8 ; per wave: single 2M x 4N pass ---
    {
      const int nt0 = wv * 4;
      v8f c0[4], c1[4];
      const __bf16* pB[4];
#pragma unroll
      for (int q = 0; q < 4; ++q) {
        const float bv = b2g[(nt0 + q) * 16 + l15];
        c0[q] = (v8f){bv, bv, bv, bv, bv, bv, bv, bv};
        c1[q] = c0[q];
        pB[q] = pW2 + (size_t)(nt0 + q) * KT2 * 512 + lane * 16;
      }
#pragma unroll 4
      for (int kt = 0; kt < KT2; ++kt) {
        const int k0 = kt * 32;
        AF a0, a1;
        a0.h[0] = *(const v8bf*)&hb[l15 * SH + k0 + half * 8];
        a0.h[1] = *(const v8bf*)&hb[l15 * SH + k0 + 16 + half * 8];
        a1.h[0] = *(const v8bf*)&hb[(16 + l15) * SH + k0 + half * 8];
        a1.h[1] = *(const v8bf*)&hb[(16 + l15) * SH + k0 + 16 + half * 8];
        AF f[4];
#pragma unroll
        for (int q = 0; q < 4; ++q)
          f[q].v = *(const v16bf*)(pB[q] + (size_t)kt * 512);
#pragma unroll
        for (int q = 0; q < 4; ++q) {
          c0[q] = WMMA_BF16(a0.v, f[q].v, c0[q]);
          c1[q] = WMMA_BF16(a1.v, f[q].v, c1[q]);
        }
      }
#pragma unroll
      for (int q = 0; q < 4; ++q) {
        const int nc = (nt0 + q) * 16 + l15;
#pragma unroll
        for (int r = 0; r < 8; ++r) {
          const int m0 = 8 * half + r;
          xf[m0 * SX + nc]        -= c0[q][r] * 0.125f;
          xf[(16 + m0) * SX + nc] -= c1[q][r] * 0.125f;
        }
      }
    }
    __syncthreads();

    // --- masked squared-error partial for this step ---
    float acc = 0.0f;
    for (int i = tid; i < MB * Dn; i += 256) {
      int m = i >> 9, c = i & (Dn - 1);
      if (t0 + m < len) {
        float d = xf[m * SX + c] - tg[i];
        acc += d * d;
      }
    }
    atomicAdd(&lsum[s], acc);
    // no barrier needed: lsum[s] only read after the final __syncthreads below,
    // and next-iteration writes touch xb (not read here) / xf (written only
    // after two more barriers).
  }
  __syncthreads();
  if (tid < STEPSn) atomicAdd(&accum[tid], lsum[tid]);
}

extern "C" void kernel_launch(void* const* d_in, const int* in_sizes, int n_in,
                              void* d_out, int out_size, void* d_ws, size_t ws_size,
                              hipStream_t stream) {
  const float* student = (const float*)d_in[0];
  const float* teacher = (const float*)d_in[1];
  const int*   lengths = (const int*)d_in[2];
  const float* wt      = (const float*)d_in[3];
  const float* bt      = (const float*)d_in[4];
  const float* W1      = (const float*)d_in[5];
  const float* b1      = (const float*)d_in[6];
  const float* W2      = (const float*)d_in[7];
  const float* b2      = (const float*)d_in[8];
  float* out = (float*)d_out;

  char* ws = (char*)d_ws;
  __bf16* pW1  = (__bf16*)ws;                                   // 512*1024*2 = 1 MB
  __bf16* pW2  = (__bf16*)(ws + (1 << 20));                     // 1024*512*2 = 1 MB
  float*  bias1 = (float*)(ws + (2 << 20));                     // 8*1024*4 = 32 KB
  float*  accum = (float*)(ws + (2 << 20) + STEPSn * Hn * 4);   // 9 floats

  hipLaunchKernelGGL(init_kernel, dim3(1), dim3(32), 0, stream, lengths, accum);
  hipLaunchKernelGGL(pack_w_kernel, dim3((Dn * Hn) / 256), dim3(256), 0, stream,
                     W1, pW1, Dn, Hn);
  hipLaunchKernelGGL(pack_w_kernel, dim3((Hn * Dn) / 256), dim3(256), 0, stream,
                     W2, pW2, Hn, Dn);
  hipLaunchKernelGGL(bias_kernel, dim3((STEPSn * Hn) / 256), dim3(256), 0, stream,
                     W1, b1, wt, bt, bias1);
  hipLaunchKernelGGL(flow_main, dim3((Bn * Tn) / MB), dim3(256), 0, stream,
                     student, teacher, lengths, pW1, pW2, bias1, b2, accum);
  hipLaunchKernelGGL(final_kernel, dim3(1), dim3(1), 0, stream, accum, out);
}
